// GraphAttentionTransformer_59227599011948
// MI455X (gfx1250) — matrix operand order repaired and checked by reading, not compiled, not run on hardware
//
#include <hip/hip_runtime.h>
#include <hip/hip_bf16.h>
#include <math.h>
#include <stdint.h>

// ---------------------------------------------------------------------------
// Equiformer-style GNN for MI455X (gfx1250, wave32).
// All GEMMs use v_wmma_f32_16x16x32_bf16 (bf16 in, f32 accumulate).
// A-tiles are staged global->LDS with GLOBAL_LOAD_ASYNC_TO_LDS_B128 (ASYNCcnt).
// edge_dst == repeat(arange(N), DEG) in the reference, so node n owns edges
// [n*32, n*32+32) -> all segment ops are per-node reductions over 32
// contiguous edges (no atomics).
// ---------------------------------------------------------------------------

static constexpr int NNODE  = 8000;
static constexpr int DEG    = 32;
static constexpr int NEDGE  = NNODE * DEG;   // 256000
static constexpr int NLAYERS = 6;

typedef __bf16 bf16;
typedef __bf16 bf16x16 __attribute__((ext_vector_type(16)));
typedef __bf16 bf16x8  __attribute__((ext_vector_type(8)));
typedef float  f32x8   __attribute__((ext_vector_type(8)));

// fast 1/x via v_rcp_f32 (bf16 matmuls already bound accuracy; avoid the
// precise-division Newton chains seen in round-1 asm)
__device__ inline float fast_rcp(float x) { return __builtin_amdgcn_rcpf(x); }
__device__ inline float fast_sigmoid(float x) {
  return fast_rcp(1.f + __expf(-x));
}

// ------------------------------ reductions ---------------------------------
__device__ inline float warp_sum(float v) {
#pragma unroll
  for (int o = 16; o > 0; o >>= 1) v += __shfl_xor(v, o, 32);
  return v;
}
__device__ inline float warp_max(float v) {
#pragma unroll
  for (int o = 16; o > 0; o >>= 1) v = fmaxf(v, __shfl_xor(v, o, 32));
  return v;
}
__device__ inline float block_sum(float v, float* sh) {
  v = warp_sum(v);
  const int w  = threadIdx.x >> 5;
  const int nw = blockDim.x >> 5;
  if ((threadIdx.x & 31) == 0) sh[w] = v;
  __syncthreads();
  float r = 0.f;
  for (int i = 0; i < nw; ++i) r += sh[i];
  __syncthreads();
  return r;
}

// ------------------------------ WMMA GEMM ----------------------------------
// C[M x NOUT] = act(A[M x K](bf16) @ B[K x NOUT](bf16) + bias)
// block = 256 threads = 8 waves as 4(M) x 2(N); block tile 64x32; K step 32.
template <int K, int NOUT, int ACT, bool OUTBF, bool BIAS>
__global__ __launch_bounds__(256) void k_gemm(const bf16* __restrict__ A,
                                              const bf16* __restrict__ B,
                                              const float* __restrict__ bias,
                                              float* __restrict__ Cf,
                                              bf16* __restrict__ Cb, int M) {
  __shared__ bf16 lA[64][40];   // row stride 80B: 16B aligned, conflict-free
  __shared__ bf16 lBt[32][40];  // B stored transposed: lBt[n][k]

  const int tid   = threadIdx.x;
  const int wave  = tid >> 5, lane = tid & 31;
  const int wm    = wave >> 1, wn = wave & 1;
  const int lrow  = lane & 15, lhalf = lane >> 4;
  const int m0    = blockIdx.x * 64;
  const int n0    = blockIdx.y * 32;

  f32x8 acc = {0.f, 0.f, 0.f, 0.f, 0.f, 0.f, 0.f, 0.f};

  const int arow_ld = tid >> 2;          // 0..63
  const int acol_ld = (tid & 3) * 8;     // 0,8,16,24
  const int brow_ld = tid >> 3;          // 0..31
  const int bcol_ld = (tid & 7) * 4;     // 0..28

  // per-thread A source pointer and LDS destination byte-address
  const bf16* aptr = &A[(size_t)(m0 + arow_ld) * K + acol_ld];
  const unsigned ldsA = (unsigned)(uintptr_t)&lA[arow_ld][acol_ld];

  for (int k0 = 0; k0 < K; k0 += 32) {
    // stage A tile (64x32): async direct-to-LDS, 16B per lane (ASYNCcnt)
    {
      const void* ga = (const void*)(aptr + k0);
      asm volatile("global_load_async_to_lds_b128 %0, %1, off"
                   :: "v"(ldsA), "v"(ga) : "memory");
    }
    // stage B tile (32x32) transposed into lBt (needs VGPR scatter anyway)
    {
      const uint2 v = *(const uint2*)&B[(size_t)(k0 + brow_ld) * NOUT + n0 + bcol_ld];
      const bf16* pv = (const bf16*)&v;
#pragma unroll
      for (int j = 0; j < 4; ++j) lBt[bcol_ld + j][brow_ld] = pv[j];
    }
    asm volatile("s_wait_asynccnt 0" ::: "memory");
    __syncthreads();

    bf16x16 af, bfr;
    {
      const int ar = wm * 16 + lrow;
      const bf16x8 a0 = *(const bf16x8*)&lA[ar][lhalf * 8];
      const bf16x8 a1 = *(const bf16x8*)&lA[ar][16 + lhalf * 8];
      const int br = wn * 16 + lrow;
      const bf16x8 b0 = *(const bf16x8*)&lBt[br][lhalf * 8];
      const bf16x8 b1 = *(const bf16x8*)&lBt[br][16 + lhalf * 8];
#pragma unroll
      for (int i = 0; i < 8; ++i) {
        af[i] = a0[i]; af[i + 8] = a1[i];
        bfr[i] = b0[i]; bfr[i + 8] = b1[i];
      }
    }
    acc = __builtin_amdgcn_wmma_f32_16x16x32_bf16(false, af, false, bfr,
                                                  (short)0, acc, false, false);
    __syncthreads();
  }

  const int n = n0 + wn * 16 + lrow;
#pragma unroll
  for (int r = 0; r < 8; ++r) {
    const int m = m0 + wm * 16 + lhalf * 8 + r;
    float v = acc[r];
    if (BIAS) v += bias[n];
    if (ACT == 1) v = v * fast_sigmoid(v);  // SiLU via v_rcp_f32
    if (OUTBF) Cb[(size_t)m * NOUT + n] = (bf16)v;
    else       Cf[(size_t)m * NOUT + n] = v;
  }
}

// --------------------------- elementwise kernels ---------------------------
__global__ void k_convert_bf16(const float* __restrict__ in,
                               bf16* __restrict__ out, int n) {
  int i = blockIdx.x * 256 + threadIdx.x;
  if (i < n) out[i] = (bf16)in[i];
}

// one wave per edge: distance, unit vector, Y1(3), Y2(5), rbf(128, bf16)
__global__ void k_edge_geom(const float* __restrict__ pos,
                            const int* __restrict__ src,
                            float* __restrict__ Y1, float* __restrict__ Y2,
                            bf16* __restrict__ rbf) {
  const int e = blockIdx.x * 8 + (threadIdx.x >> 5);
  const int lane = threadIdx.x & 31;
  if (e >= NEDGE) return;
  const int s = src[e];
  const int t = e >> 5;  // dst = e / DEG (contiguous pattern)
  const float vx = pos[s * 3 + 0] - pos[t * 3 + 0];
  const float vy = pos[s * 3 + 1] - pos[t * 3 + 1];
  const float vz = pos[s * 3 + 2] - pos[t * 3 + 2];
  const float d = sqrtf(vx * vx + vy * vy + vz * vz + 1e-12f);
  const float iv = fast_rcp(d);
  const float x = vx * iv, y = vy * iv, z = vz * iv;
  if (lane == 0) {
    const float s3 = 1.7320508075688772f;   // sqrt(3)
    const float c15 = 3.872983346207417f;   // sqrt(15)
    const float c5h = 1.118033988749895f;   // sqrt(5)/2
    Y1[e * 3 + 0] = s3 * x; Y1[e * 3 + 1] = s3 * y; Y1[e * 3 + 2] = s3 * z;
    Y2[e * 5 + 0] = c15 * x * y;
    Y2[e * 5 + 1] = c15 * y * z;
    Y2[e * 5 + 2] = c5h * (3.f * z * z - 1.f);
    Y2[e * 5 + 3] = c15 * x * z;
    Y2[e * 5 + 4] = 0.5f * c15 * (x * x - y * y);
  }
  const float sig_inv = 128.f / 5.f;  // 1/sigma
#pragma unroll
  for (int j4 = 0; j4 < 4; ++j4) {
    const int j = lane * 4 + j4;
    const float mu = 5.f * (float)j / 127.f;
    const float tt = (d - mu) * sig_inv;
    rbf[(size_t)e * 128 + j] = (bf16)__expf(-0.5f * tt * tt);
  }
}

// x0 = emb[atom] + inv*segsum(w[:, :128]); x1/x2 = inv*segsum(w * Y)
__global__ void k_node_init(const int* __restrict__ atom,
                            const float* __restrict__ emb,
                            const float* __restrict__ w,
                            const float* __restrict__ Y1,
                            const float* __restrict__ Y2,
                            float* __restrict__ x0, float* __restrict__ x1,
                            float* __restrict__ x2) {
  const int n = blockIdx.x;
  const float inv = 0.04472135954999579f;  // 1/sqrt(500)
  for (int ch = threadIdx.x; ch < 480; ch += blockDim.x) {
    float acc = 0.f;
    if (ch < 128) {
      for (int i = 0; i < DEG; ++i) acc += w[(size_t)(n * DEG + i) * 224 + ch];
      x0[n * 128 + ch] = emb[atom[n] * 128 + ch] + inv * acc;
    } else if (ch < 320) {
      const int idx = ch - 128, c = idx / 3, dd = idx % 3;
      for (int i = 0; i < DEG; ++i) {
        const int e = n * DEG + i;
        acc += w[(size_t)e * 224 + 128 + c] * Y1[e * 3 + dd];
      }
      x1[n * 192 + idx] = inv * acc;
    } else {
      const int idx = ch - 320, c = idx / 5, dd = idx % 5;
      for (int i = 0; i < DEG; ++i) {
        const int e = n * DEG + i;
        acc += w[(size_t)e * 224 + 192 + c] * Y2[e * 5 + dd];
      }
      x2[n * 160 + idx] = inv * acc;
    }
  }
}

// eq_norm: layernorm(x0) -> f32 + bf16 copy; x1/x2 scaled by RMS over channels
__global__ void k_eqnorm(const float* __restrict__ x0,
                         const float* __restrict__ x1,
                         const float* __restrict__ x2,
                         float* __restrict__ x0n, bf16* __restrict__ x0nb,
                         float* __restrict__ x1n, float* __restrict__ x2n) {
  __shared__ float sh[8];
  const int n = blockIdx.x, t = threadIdx.x;  // 128 threads
  const float v = x0[n * 128 + t];
  const float s  = block_sum(v, sh);
  const float s2 = block_sum(v * v, sh);
  const float mean = s / 128.f;
  const float var  = s2 / 128.f - mean * mean;
  const float rs   = rsqrtf(var + 1e-5f);
  const float o = (v - mean) * rs;
  x0n[n * 128 + t] = o; x0nb[n * 128 + t] = (bf16)o;

  float p1 = 0.f;
  for (int i = t; i < 192; i += 128) { float a = x1[n * 192 + i]; p1 += a * a; }
  const float rn1 = rsqrtf(block_sum(p1, sh) / 64.f + 1e-5f);
  for (int i = t; i < 192; i += 128) x1n[n * 192 + i] = x1[n * 192 + i] * rn1;

  float p2 = 0.f;
  for (int i = t; i < 160; i += 128) { float a = x2[n * 160 + i]; p2 += a * a; }
  const float rn2 = rsqrtf(block_sum(p2, sh) / 32.f + 1e-5f);
  for (int i = t; i < 160; i += 128) x2n[n * 160 + i] = x2[n * 160 + i] * rn2;
}

// s1[c] = sum_d x1n[src][c][d]*Y1[d]; s2 similar -> s12 (E x 96, bf16)
__global__ void k_edge_s(const int* __restrict__ src,
                         const float* __restrict__ x1n,
                         const float* __restrict__ x2n,
                         const float* __restrict__ Y1,
                         const float* __restrict__ Y2,
                         bf16* __restrict__ s12) {
  const long long idx = (long long)blockIdx.x * blockDim.x + threadIdx.x;
  if (idx >= (long long)NEDGE * 96) return;
  const int e = (int)(idx / 96), ch = (int)(idx % 96);
  const int s = src[e];
  float acc = 0.f;
  if (ch < 64) {
#pragma unroll
    for (int dd = 0; dd < 3; ++dd) acc += x1n[s * 192 + ch * 3 + dd] * Y1[e * 3 + dd];
  } else {
    const int c = ch - 64;
#pragma unroll
    for (int dd = 0; dd < 5; ++dd) acc += x2n[s * 160 + c * 5 + dd] * Y2[e * 5 + dd];
  }
  s12[(size_t)e * 96 + ch] = (bf16)acc;
}

// v0 = g0 * (x0n[src] + p); logits[e][h] = q[dst,h,:].v0[h,:]/sqrt(32)
__global__ void k_edge_v0(const float* __restrict__ g,
                          const float* __restrict__ p,
                          const float* __restrict__ x0n,
                          const int* __restrict__ src,
                          const float* __restrict__ q,
                          float* __restrict__ v0, float* __restrict__ logits) {
  const int e = blockIdx.x, c = threadIdx.x;  // 128 threads, wave == head
  const int s = src[e], dn = e >> 5;
  const float val = g[(size_t)e * 320 + c] * (x0n[s * 128 + c] + p[(size_t)e * 128 + c]);
  v0[(size_t)e * 128 + c] = val;
  float pr = warp_sum(val * q[dn * 128 + c]);
  if ((c & 31) == 0) logits[e * 4 + (c >> 5)] = pr * 0.17677669529663687f;
}

// softmax over the 32 contiguous edges of each node, per head
__global__ void k_softmax(const float* __restrict__ logits,
                          float* __restrict__ alpha) {
  const int n = blockIdx.x;
  const int lane = threadIdx.x & 31, h = threadIdx.x >> 5;  // 128 threads
  const int e = n * DEG + lane;
  const float v = logits[e * 4 + h];
  const float m = warp_max(v);
  const float ex = __expf(v - m);
  const float den = warp_sum(ex);
  alpha[e * 4 + h] = ex * fast_rcp(den + 1e-9f);
}

// a0 = sum alpha*v0 (bf16 out); a1/a2 with v1/v2 built on the fly
__global__ void k_aggregate(const float* __restrict__ alpha,
                            const float* __restrict__ v0,
                            const float* __restrict__ g,
                            const int* __restrict__ src,
                            const float* __restrict__ x1n,
                            const float* __restrict__ x2n,
                            const float* __restrict__ xq1,
                            const float* __restrict__ xq2,
                            const float* __restrict__ Y1,
                            const float* __restrict__ Y2,
                            bf16* __restrict__ a0b, float* __restrict__ a1,
                            float* __restrict__ a2) {
  const int n = blockIdx.x;
  for (int ch = threadIdx.x; ch < 480; ch += blockDim.x) {
    float acc = 0.f;
    if (ch < 128) {
      const int h = ch >> 5;
      for (int i = 0; i < DEG; ++i) {
        const int e = n * DEG + i;
        acc += alpha[e * 4 + h] * v0[(size_t)e * 128 + ch];
      }
      a0b[n * 128 + ch] = (bf16)acc;
    } else if (ch < 320) {
      const int idx = ch - 128, c = idx / 3, dd = idx % 3, h = c >> 4;  // D1/NH=16
      for (int i = 0; i < DEG; ++i) {
        const int e = n * DEG + i, s = src[e];
        const float v1 = g[(size_t)e * 320 + 128 + c] * x1n[s * 192 + idx] +
                         g[(size_t)e * 320 + 224 + c] * xq1[s * 64 + c] * Y1[e * 3 + dd];
        acc += alpha[e * 4 + h] * v1;
      }
      a1[n * 192 + idx] = acc;
    } else {
      const int idx = ch - 320, c = idx / 5, dd = idx % 5, h = c >> 3;  // D2/NH=8
      for (int i = 0; i < DEG; ++i) {
        const int e = n * DEG + i, s = src[e];
        const float v2 = g[(size_t)e * 320 + 192 + c] * x2n[s * 160 + idx] +
                         g[(size_t)e * 320 + 288 + c] * xq2[s * 32 + c] * Y2[e * 5 + dd];
        acc += alpha[e * 4 + h] * v2;
      }
      a2[n * 160 + idx] = acc;
    }
  }
}

// x0 += a0@Wo0 (t0 precomputed); x1 += a1@Wo1; x2 += a2@Wo2
__global__ void k_update1(const float* __restrict__ t0,
                          const float* __restrict__ a1,
                          const float* __restrict__ a2,
                          const float* __restrict__ Wo1,
                          const float* __restrict__ Wo2,
                          float* __restrict__ x0, float* __restrict__ x1,
                          float* __restrict__ x2) {
  const int n = blockIdx.x;
  for (int ch = threadIdx.x; ch < 480; ch += blockDim.x) {
    if (ch < 128) {
      x0[n * 128 + ch] += t0[n * 128 + ch];
    } else if (ch < 320) {
      const int idx = ch - 128, c = idx / 3, dd = idx % 3;
      float acc = 0.f;
      for (int cc = 0; cc < 64; ++cc) acc += a1[n * 192 + cc * 3 + dd] * Wo1[cc * 64 + c];
      x1[n * 192 + idx] += acc;
    } else {
      const int idx = ch - 320, c = idx / 5, dd = idx % 5;
      float acc = 0.f;
      for (int cc = 0; cc < 32; ++cc) acc += a2[n * 160 + cc * 5 + dd] * Wo2[cc * 32 + c];
      x2[n * 160 + idx] += acc;
    }
  }
}

// x0 += f[:,:128]; x1 += sigmoid(f1)*(x1n@Wv1); x2 += sigmoid(f2)*(x2n@Wv2)
__global__ void k_ffn_update(const float* __restrict__ f,
                             const float* __restrict__ x1n,
                             const float* __restrict__ x2n,
                             const float* __restrict__ Wv1,
                             const float* __restrict__ Wv2,
                             float* __restrict__ x0, float* __restrict__ x1,
                             float* __restrict__ x2) {
  const int n = blockIdx.x;
  for (int ch = threadIdx.x; ch < 480; ch += blockDim.x) {
    if (ch < 128) {
      x0[n * 128 + ch] += f[(size_t)n * 224 + ch];
    } else if (ch < 320) {
      const int idx = ch - 128, c = idx / 3, dd = idx % 3;
      const float gate = fast_sigmoid(f[(size_t)n * 224 + 128 + c]);
      float acc = 0.f;
      for (int cc = 0; cc < 64; ++cc) acc += x1n[n * 192 + cc * 3 + dd] * Wv1[cc * 64 + c];
      x1[n * 192 + idx] += gate * acc;
    } else {
      const int idx = ch - 320, c = idx / 5, dd = idx % 5;
      const float gate = fast_sigmoid(f[(size_t)n * 224 + 192 + c]);
      float acc = 0.f;
      for (int cc = 0; cc < 32; ++cc) acc += x2n[n * 160 + cc * 5 + dd] * Wv2[cc * 32 + c];
      x2[n * 160 + idx] += gate * acc;
    }
  }
}

// invfeat = [x0, |x1|_d, |x2|_d] -> bf16 (N x 224)
__global__ void k_invfeat(const float* __restrict__ x0,
                          const float* __restrict__ x1,
                          const float* __restrict__ x2,
                          bf16* __restrict__ out) {
  const int n = blockIdx.x;
  for (int ch = threadIdx.x; ch < 224; ch += blockDim.x) {
    float v;
    if (ch < 128) {
      v = x0[n * 128 + ch];
    } else if (ch < 192) {
      const int c = ch - 128; float s = 0.f;
#pragma unroll
      for (int dd = 0; dd < 3; ++dd) { float a = x1[n * 192 + c * 3 + dd]; s += a * a; }
      v = sqrtf(s + 1e-12f);
    } else {
      const int c = ch - 192; float s = 0.f;
#pragma unroll
      for (int dd = 0; dd < 5; ++dd) { float a = x2[n * 160 + c * 5 + dd]; s += a * a; }
      v = sqrtf(s + 1e-12f);
    }
    out[(size_t)n * 224 + ch] = (bf16)v;
  }
}

// LayerNorm over 512 with affine, bf16 out (256 threads, 2 elems each)
__global__ void k_ln_head(const float* __restrict__ f,
                          const float* __restrict__ gw,
                          const float* __restrict__ bw, bf16* __restrict__ out) {
  __shared__ float sh[8];
  const int n = blockIdx.x, t = threadIdx.x;
  const float v0 = f[(size_t)n * 512 + t];
  const float v1 = f[(size_t)n * 512 + t + 256];
  const float s  = block_sum(v0 + v1, sh);
  const float s2 = block_sum(v0 * v0 + v1 * v1, sh);
  const float mean = s / 512.f;
  const float var  = s2 / 512.f - mean * mean;
  const float rs   = rsqrtf(var + 1e-5f);
  out[(size_t)n * 512 + t]       = (bf16)((v0 - mean) * rs * gw[t] + bw[t]);
  out[(size_t)n * 512 + t + 256] = (bf16)((v1 - mean) * rs * gw[t + 256] + bw[t + 256]);
}

// ------------------------------- host side ---------------------------------
template <int K, int NOUT, int ACT, bool OUTBF, bool BIAS>
static void gemm(const bf16* A, const bf16* B, const float* bias, void* C,
                 int M, hipStream_t st) {
  dim3 g(M / 64, NOUT / 32), b(256);
  k_gemm<K, NOUT, ACT, OUTBF, BIAS><<<g, b, 0, st>>>(
      A, B, bias, OUTBF ? nullptr : (float*)C, OUTBF ? (bf16*)C : nullptr, M);
}

extern "C" void kernel_launch(void* const* d_in, const int* in_sizes, int n_in,
                              void* d_out, int out_size, void* d_ws,
                              size_t ws_size, hipStream_t stream) {
  (void)in_sizes; (void)n_in; (void)out_size; (void)ws_size;
  // ---- inputs (setup_inputs order; params flattened in jax pytree order:
  //      dict keys sorted: Wh1,Wh2,Wout,bh1,bh2,deg_W1..3,emb,layers,ln_b,ln_g;
  //      each layer dict sorted: P10,P20,Q01,Q02,We1..3,Wf1,Wf2,Wo0..2,Wq,Wv1,Wv2)
  const float* pos[2]  = {(const float*)d_in[1], (const float*)d_in[6]};
  const int*   atom[2] = {(const int*)d_in[3], (const int*)d_in[8]};
  const int*   esrc[2] = {(const int*)d_in[4], (const int*)d_in[9]};
  const float* Wh1  = (const float*)d_in[11];
  const float* Wh2  = (const float*)d_in[12];
  const float* Wout = (const float*)d_in[13];
  const float* bh1  = (const float*)d_in[14];
  const float* bh2  = (const float*)d_in[15];
  const float* degW1 = (const float*)d_in[16];
  const float* degW2 = (const float*)d_in[17];
  const float* degW3 = (const float*)d_in[18];
  const float* emb   = (const float*)d_in[19];
  const float* lnb = (const float*)d_in[110];
  const float* lng = (const float*)d_in[111];
  struct LP { const float *P10,*P20,*Q01,*Q02,*We1,*We2,*We3,*Wf1,*Wf2,*Wo0,*Wo1,*Wo2,*Wq,*Wv1,*Wv2; };
  LP lp[NLAYERS];
  for (int l = 0; l < NLAYERS; ++l) {
    const int b0 = 20 + l * 15;
    lp[l] = {(const float*)d_in[b0+0],(const float*)d_in[b0+1],(const float*)d_in[b0+2],
             (const float*)d_in[b0+3],(const float*)d_in[b0+4],(const float*)d_in[b0+5],
             (const float*)d_in[b0+6],(const float*)d_in[b0+7],(const float*)d_in[b0+8],
             (const float*)d_in[b0+9],(const float*)d_in[b0+10],(const float*)d_in[b0+11],
             (const float*)d_in[b0+12],(const float*)d_in[b0+13],(const float*)d_in[b0+14]};
  }

  // ---- workspace bump allocator
  char* wsp = (char*)d_ws;
  auto alloc = [&](size_t bytes) -> void* {
    void* p = wsp; wsp += (bytes + 255) & ~(size_t)255; return p;
  };
  auto allocF = [&](size_t n) { return (float*)alloc(n * 4); };
  auto allocB = [&](size_t n) { return (bf16*)alloc(n * 2); };

  // bf16 weights
  bf16* wbDeg1 = allocB(128 * 64);
  bf16* wbDeg2 = allocB(64 * 64);
  bf16* wbDeg3 = allocB(64 * 224);
  bf16* wbWout = allocB(224 * 512);
  bf16* wbWh1  = allocB(512 * 512);
  bf16* wbWh2  = allocB(512 * 128);
  struct LW { bf16 *We1,*We2,*We3,*P,*Q01,*Q02,*Wq,*Wo0,*Wf1,*Wf2; };
  LW lw[NLAYERS];
  for (int l = 0; l < NLAYERS; ++l) {
    lw[l].We1 = allocB(128 * 64);  lw[l].We2 = allocB(64 * 64);
    lw[l].We3 = allocB(64 * 320);  lw[l].P   = allocB(96 * 128);
    lw[l].Q01 = allocB(128 * 64);  lw[l].Q02 = allocB(128 * 32);
    lw[l].Wq  = allocB(128 * 128); lw[l].Wo0 = allocB(128 * 128);
    lw[l].Wf1 = allocB(128 * 128); lw[l].Wf2 = allocB(128 * 224);
  }
  // activations (reused across both branches)
  bf16*  rbf  = allocB((size_t)NEDGE * 128);
  float* Y1   = allocF((size_t)NEDGE * 3);
  float* Y2   = allocF((size_t)NEDGE * 5);
  bf16*  eh1  = allocB((size_t)NEDGE * 64);
  bf16*  eh2  = allocB((size_t)NEDGE * 64);
  float* gbuf = allocF((size_t)NEDGE * 320);  // also reused for deg-MLP output (E x 224)
  bf16*  s12  = allocB((size_t)NEDGE * 96);
  float* pbuf = allocF((size_t)NEDGE * 128);
  float* v0   = allocF((size_t)NEDGE * 128);
  float* logi = allocF((size_t)NEDGE * 4);
  float* alph = allocF((size_t)NEDGE * 4);
  float* x0 = allocF((size_t)NNODE * 128);
  float* x1 = allocF((size_t)NNODE * 192);
  float* x2 = allocF((size_t)NNODE * 160);
  float* x0n = allocF((size_t)NNODE * 128);
  bf16*  x0nb = allocB((size_t)NNODE * 128);
  float* x1n = allocF((size_t)NNODE * 192);
  float* x2n = allocF((size_t)NNODE * 160);
  float* qb  = allocF((size_t)NNODE * 128);
  float* xq1 = allocF((size_t)NNODE * 64);
  float* xq2 = allocF((size_t)NNODE * 32);
  bf16*  a0b = allocB((size_t)NNODE * 128);
  float* a1  = allocF((size_t)NNODE * 192);
  float* a2  = allocF((size_t)NNODE * 160);
  float* t0  = allocF((size_t)NNODE * 128);
  bf16*  tf  = allocB((size_t)NNODE * 128);
  float* fbuf = allocF((size_t)NNODE * 224);
  bf16*  invb = allocB((size_t)NNODE * 224);
  float* fh   = allocF((size_t)NNODE * 512);
  bf16*  fnb  = allocB((size_t)NNODE * 512);
  bf16*  hb   = allocB((size_t)NNODE * 512);

  auto conv = [&](const float* s, bf16* dst, int n) {
    k_convert_bf16<<<(n + 255) / 256, 256, 0, stream>>>(s, dst, n);
  };

  // ---- convert weights once per launch
  conv(degW1, wbDeg1, 128 * 64); conv(degW2, wbDeg2, 64 * 64);
  conv(degW3, wbDeg3, 64 * 224);
  conv(Wout, wbWout, 224 * 512); conv(Wh1, wbWh1, 512 * 512);
  conv(Wh2, wbWh2, 512 * 128);
  for (int l = 0; l < NLAYERS; ++l) {
    conv(lp[l].We1, lw[l].We1, 128 * 64); conv(lp[l].We2, lw[l].We2, 64 * 64);
    conv(lp[l].We3, lw[l].We3, 64 * 320);
    conv(lp[l].P10, lw[l].P, 64 * 128);                 // rows 0..63
    conv(lp[l].P20, lw[l].P + 64 * 128, 32 * 128);      // rows 64..95
    conv(lp[l].Q01, lw[l].Q01, 128 * 64); conv(lp[l].Q02, lw[l].Q02, 128 * 32);
    conv(lp[l].Wq, lw[l].Wq, 128 * 128);  conv(lp[l].Wo0, lw[l].Wo0, 128 * 128);
    conv(lp[l].Wf1, lw[l].Wf1, 128 * 128); conv(lp[l].Wf2, lw[l].Wf2, 128 * 224);
  }

  // ---- two branches, shared params
  for (int b = 0; b < 2; ++b) {
    float* outp = (float*)d_out + (size_t)b * NNODE * 128;

    k_edge_geom<<<NEDGE / 8, 256, 0, stream>>>(pos[b], esrc[b], Y1, Y2, rbf);

    // degree MLP: w = mlp3(rbf) -> (E x 224) into gbuf
    gemm<128, 64, 1, true, false>(rbf, wbDeg1, nullptr, eh1, NEDGE, stream);
    gemm<64, 64, 1, true, false>(eh1, wbDeg2, nullptr, eh2, NEDGE, stream);
    gemm<64, 224, 0, false, false>(eh2, wbDeg3, nullptr, gbuf, NEDGE, stream);
    k_node_init<<<NNODE, 256, 0, stream>>>(atom[b], emb, gbuf, Y1, Y2, x0, x1, x2);

    for (int l = 0; l < NLAYERS; ++l) {
      k_eqnorm<<<NNODE, 128, 0, stream>>>(x0, x1, x2, x0n, x0nb, x1n, x2n);
      gemm<128, 64, 0, false, false>(x0nb, lw[l].Q01, nullptr, xq1, NNODE, stream);
      gemm<128, 32, 0, false, false>(x0nb, lw[l].Q02, nullptr, xq2, NNODE, stream);
      gemm<128, 128, 0, false, false>(x0nb, lw[l].Wq, nullptr, qb, NNODE, stream);

      gemm<128, 64, 1, true, false>(rbf, lw[l].We1, nullptr, eh1, NEDGE, stream);
      gemm<64, 64, 1, true, false>(eh1, lw[l].We2, nullptr, eh2, NEDGE, stream);
      gemm<64, 320, 0, false, false>(eh2, lw[l].We3, nullptr, gbuf, NEDGE, stream);

      k_edge_s<<<((size_t)NEDGE * 96 + 255) / 256, 256, 0, stream>>>(
          esrc[b], x1n, x2n, Y1, Y2, s12);
      gemm<96, 128, 0, false, false>(s12, lw[l].P, nullptr, pbuf, NEDGE, stream);
      k_edge_v0<<<NEDGE, 128, 0, stream>>>(gbuf, pbuf, x0n, esrc[b], qb, v0, logi);
      k_softmax<<<NNODE, 128, 0, stream>>>(logi, alph);
      k_aggregate<<<NNODE, 256, 0, stream>>>(alph, v0, gbuf, esrc[b], x1n, x2n,
                                             xq1, xq2, Y1, Y2, a0b, a1, a2);
      gemm<128, 128, 0, false, false>(a0b, lw[l].Wo0, nullptr, t0, NNODE, stream);
      k_update1<<<NNODE, 256, 0, stream>>>(t0, a1, a2, lp[l].Wo1, lp[l].Wo2, x0, x1, x2);

      k_eqnorm<<<NNODE, 128, 0, stream>>>(x0, x1, x2, x0n, x0nb, x1n, x2n);
      gemm<128, 128, 1, true, false>(x0nb, lw[l].Wf1, nullptr, tf, NNODE, stream);
      gemm<128, 224, 0, false, false>(tf, lw[l].Wf2, nullptr, fbuf, NNODE, stream);
      k_ffn_update<<<NNODE, 256, 0, stream>>>(fbuf, x1n, x2n, lp[l].Wv1, lp[l].Wv2,
                                              x0, x1, x2);
    }

    // head
    k_invfeat<<<NNODE, 256, 0, stream>>>(x0, x1, x2, invb);
    gemm<224, 512, 0, false, false>(invb, wbWout, nullptr, fh, NNODE, stream);
    k_ln_head<<<NNODE, 256, 0, stream>>>(fh, lng, lnb, fnb);
    gemm<512, 512, 1, true, true>(fnb, wbWh1, bh1, hb, NNODE, stream);
    gemm<512, 128, 0, false, true>(hb, wbWh2, bh2, outp, NNODE, stream);
  }
}
